// MultiHeadedAttention_41583873359904
// MI455X (gfx1250) — matrix operand order repaired and checked by reading, not compiled, not run on hardware
//
#include <hip/hip_runtime.h>
#include <cstdint>
#include <cstddef>

// ---------------------------------------------------------------------------
// Shapes (fixed by the reference)
// ---------------------------------------------------------------------------
#define BATCH 8
#define CCH   256
#define HH    128
#define WW    128
#define HWSZ  (HH * WW)                      // 16384
#define NELEM ((size_t)BATCH * CCH * HWSZ)   // 33,554,432
#define BN_EPS 1e-5f
#define LEAKY  0.2f

// ---------------------------------------------------------------------------
// WMMA types (CDNA5 gfx1250, wave32): v_wmma_f32_16x16x32_bf16
// ---------------------------------------------------------------------------
typedef __attribute__((ext_vector_type(16))) __bf16 v16bf;
typedef __attribute__((ext_vector_type(8)))  __bf16 v8bf;
typedef __attribute__((ext_vector_type(8)))  float  v8f;

// Hardware f32 -> bf16 (backend selects v_cvt_pk_bf16_f32-class ops)
__device__ __forceinline__ __bf16 f2bf(float f) { return (__bf16)f; }

// ---------------------------------------------------------------------------
// CDNA5 async global->LDS copy (pure byte copy path, tracked by ASYNCcnt).
// One instruction moves 16B per active lane (512B/wave).
// ---------------------------------------------------------------------------
__device__ __forceinline__ void async_copy_b128(const void* gptr, void* lptr) {
    uint32_t loff  = (uint32_t)(uintptr_t)lptr;   // low 32 bits of flat = LDS offset
    uint64_t gaddr = (uint64_t)(uintptr_t)gptr;
    asm volatile("global_load_async_to_lds_b128 %0, %1, off"
                 :: "v"(loff), "v"(gaddr) : "memory");
}
__device__ __forceinline__ void wait_async() {
    asm volatile("s_wait_asynccnt 0x0" ::: "memory");
}

// A fragment, 16x32 bf16, LDS row-major [M][K], row stride `stride` (elems).
// lanes 0-15: M=lane, K={k0..k0+7, k0+16..k0+23};
// lanes 16-31: M=lane-16, K={k0+8..k0+15, k0+24..k0+31}.
__device__ __forceinline__ v16bf frag_a(const __bf16* base, int stride,
                                        int m0, int k0, int lane) {
    int row  = m0 + (lane & 15);
    int ksel = (lane >> 4) * 8;
    union { v16bf v; v8bf h[2]; } u;
    u.h[0] = *reinterpret_cast<const v8bf*>(base + row * stride + k0 + ksel);
    u.h[1] = *reinterpret_cast<const v8bf*>(base + row * stride + k0 + 16 + ksel);
    return u.v;
}

// B fragment, 32x16 bf16, LDS stored transposed [N][K].
// lanes 0-15: N=lane, K={k0..k0+7, k0+16..k0+23}; lanes 16-31: N=lane-16, K={k0+8..k0+15, k0+24..k0+31}.
__device__ __forceinline__ v16bf frag_b(const __bf16* base, int stride,
                                        int n0, int k0, int lane) {
    int col  = n0 + (lane & 15);
    int ksel = (lane >> 4) * 8;
    const __bf16* p = base + col * stride + k0 + ksel;
    union { v16bf v; v8bf h[2]; } u;
    u.h[0] = *reinterpret_cast<const v8bf*>(p);
    u.h[1] = *reinterpret_cast<const v8bf*>(p + 16);
    return u.v;
}

#define WMMA_BF16(a, b, c) \
    __builtin_amdgcn_wmma_f32_16x16x32_bf16(false, (a), false, (b), (short)0, (c), false, false)

#define V8F_ZERO {0.f,0.f,0.f,0.f,0.f,0.f,0.f,0.f}

// C/D layout: element r -> row m0 + r + ((lane>=16)?8:0), col n0 + (lane&15).

// ---------------------------------------------------------------------------
// Kernel 1: 1x1 projection  Out[b] = W(256x256) * X[b](256x16384) + bias
// fp32 in, bf16 out. grid (HW/128, C/128, B), block 256; wave = 64(M) x 32(N).
// ---------------------------------------------------------------------------
__global__ __launch_bounds__(256) void proj1x1_kernel(
    const float* __restrict__ X, const float* __restrict__ Wt,
    const float* __restrict__ bias, __bf16* __restrict__ Out) {
    __shared__ __align__(16) __bf16 As[128 * 32];
    __shared__ __align__(16) __bf16 Bs[128 * 32];

    const int tid = threadIdx.x, lane = tid & 31, wid = tid >> 5;
    const int wave_m = wid >> 2, wave_n = wid & 3;          // 2 x 4 wave grid
    const int n_base = blockIdx.x * 128;
    const int oc0    = blockIdx.y * 128;
    const float*  Xb = X   + (size_t)blockIdx.z * CCH * HWSZ;
    __bf16*       Ob = Out + (size_t)blockIdx.z * CCH * HWSZ;

    v8f acc[4][2];
#pragma unroll
    for (int i = 0; i < 4; ++i)
#pragma unroll
        for (int j = 0; j < 2; ++j) acc[i][j] = (v8f)V8F_ZERO;

    for (int kb = 0; kb < CCH / 32; ++kb) {
        const int k_base = kb * 32;
        // A: weights 128x32 (fp32 -> bf16, hardware cvt)
#pragma unroll
        for (int t = tid; t < 1024; t += 256) {
            int row = t >> 3, kk4 = (t & 7) * 4;
            float4 w4 = *reinterpret_cast<const float4*>(
                &Wt[(size_t)(oc0 + row) * CCH + k_base + kk4]);
            As[row * 32 + kk4 + 0] = f2bf(w4.x);
            As[row * 32 + kk4 + 1] = f2bf(w4.y);
            As[row * 32 + kk4 + 2] = f2bf(w4.z);
            As[row * 32 + kk4 + 3] = f2bf(w4.w);
        }
        // B: activations 32(K) x 128(N) -> LDS transposed [N][K]
#pragma unroll
        for (int t = tid; t < 1024; t += 256) {
            int kk = t >> 5, n4 = (t & 31) * 4;
            float4 x4 = *reinterpret_cast<const float4*>(
                &Xb[(size_t)(k_base + kk) * HWSZ + n_base + n4]);
            Bs[(n4 + 0) * 32 + kk] = f2bf(x4.x);
            Bs[(n4 + 1) * 32 + kk] = f2bf(x4.y);
            Bs[(n4 + 2) * 32 + kk] = f2bf(x4.z);
            Bs[(n4 + 3) * 32 + kk] = f2bf(x4.w);
        }
        __syncthreads();

        v16bf af[4], bf[2];
#pragma unroll
        for (int i = 0; i < 4; ++i) af[i] = frag_a(As, 32, wave_m * 64 + i * 16, 0, lane);
#pragma unroll
        for (int j = 0; j < 2; ++j) bf[j] = frag_b(Bs, 32, wave_n * 32 + j * 16, 0, lane);
#pragma unroll
        for (int i = 0; i < 4; ++i)
#pragma unroll
            for (int j = 0; j < 2; ++j) acc[i][j] = WMMA_BF16(af[i], bf[j], acc[i][j]);
        __syncthreads();
    }

    const int mhi = (lane >> 4) * 8, ncol = lane & 15;
#pragma unroll
    for (int i = 0; i < 4; ++i)
#pragma unroll
        for (int j = 0; j < 2; ++j)
#pragma unroll
            for (int r = 0; r < 8; ++r) {
                int m = wave_m * 64 + i * 16 + r + mhi;
                int n = wave_n * 32 + j * 16 + ncol;
                Ob[(size_t)(oc0 + m) * HWSZ + n_base + n] =
                    f2bf(acc[i][j][r] + bias[oc0 + m]);
            }
}

// ---------------------------------------------------------------------------
// Kernel 2: attention over H per (b,c):  S = QK^T/sqrt(W); P = softmax(S);
// att = P V, written in place over V (all bf16 in HBM). grid (C, B), block 256.
// LDS: 32 + 32 + 64 = 128 KB.
// ---------------------------------------------------------------------------
__global__ __launch_bounds__(256) void attention_kernel(
    const __bf16* __restrict__ Q, const __bf16* __restrict__ K,
    __bf16* __restrict__ V /* in: v, out: att */) {
    __shared__ __align__(16) __bf16 QPs[HH * WW];  // Q, later P
    __shared__ __align__(16) __bf16 KVs[HH * WW];  // K (row-major), later V^T
    __shared__ __align__(16) float  Ss [HH * HH];  // scores

    const int tid = threadIdx.x, lane = tid & 31, wid = tid >> 5;
    const int wave_m = wid >> 1, wave_n = wid & 1;          // 4 x 2 wave grid
    const size_t base = ((size_t)blockIdx.y * CCH + blockIdx.x) * HWSZ;
    const int mhi = (lane >> 4) * 8, ncol = lane & 15;

    // Phase 1: Q, K -> LDS via CDNA5 async copy (pure bf16 byte copy)
#pragma unroll
    for (int t = tid; t < 2048; t += 256) {
        async_copy_b128(Q + base + t * 8, &QPs[t * 8]);
        async_copy_b128(K + base + t * 8, &KVs[t * 8]);
    }
    wait_async();
    __syncthreads();

    // Phase 2: S = Q K^T * scale  (K's natural [h'][w] layout == B's [N][K])
    {
        v8f acc[2][4];
#pragma unroll
        for (int i = 0; i < 2; ++i)
#pragma unroll
            for (int j = 0; j < 4; ++j) acc[i][j] = (v8f)V8F_ZERO;
        for (int kb = 0; kb < 4; ++kb) {
            int k0 = kb * 32;
            v16bf af[2], bf[4];
#pragma unroll
            for (int i = 0; i < 2; ++i) af[i] = frag_a(QPs, WW, wave_m * 32 + i * 16, k0, lane);
#pragma unroll
            for (int j = 0; j < 4; ++j) bf[j] = frag_b(KVs, WW, wave_n * 64 + j * 16, k0, lane);
#pragma unroll
            for (int i = 0; i < 2; ++i)
#pragma unroll
                for (int j = 0; j < 4; ++j) acc[i][j] = WMMA_BF16(af[i], bf[j], acc[i][j]);
        }
        const float scale = 0.08838834764831845f;  // 1/sqrt(128)
#pragma unroll
        for (int i = 0; i < 2; ++i)
#pragma unroll
            for (int j = 0; j < 4; ++j)
#pragma unroll
                for (int r = 0; r < 8; ++r) {
                    int m = wave_m * 32 + i * 16 + r + mhi;
                    int n = wave_n * 64 + j * 16 + ncol;
                    Ss[m * HH + n] = acc[i][j][r] * scale;
                }
    }
    __syncthreads();

    // Phase 3a: V -> LDS transposed [w][h'] (bf16 copy + transpose)
#pragma unroll
    for (int t = tid; t < 2048; t += 256) {
        int h = t >> 4, w8 = (t & 15) * 8;
        v8bf v8 = *reinterpret_cast<const v8bf*>(&V[base + h * WW + w8]);
#pragma unroll
        for (int u = 0; u < 8; ++u) KVs[(w8 + u) * HH + h] = v8[u];
    }
    // Phase 3b: row softmax -> P (bf16, over the dead Q buffer)
    if (tid < HH) {
        float mx = -3.4e38f;
        for (int j = 0; j < HH; ++j) mx = fmaxf(mx, Ss[tid * HH + j]);
        float s = 0.f;
        for (int j = 0; j < HH; ++j) {
            float e = __expf(Ss[tid * HH + j] - mx);
            Ss[tid * HH + j] = e;
            s += e;
        }
        float inv = 1.f / s;
        for (int j = 0; j < HH; ++j) QPs[tid * WW + j] = f2bf(Ss[tid * HH + j] * inv);
    }
    __syncthreads();

    // Phase 4: att = P V -> global bf16 (in place over V)
    {
        v8f acc[2][4];
#pragma unroll
        for (int i = 0; i < 2; ++i)
#pragma unroll
            for (int j = 0; j < 4; ++j) acc[i][j] = (v8f)V8F_ZERO;
        for (int kb = 0; kb < 4; ++kb) {
            int k0 = kb * 32;
            v16bf af[2], bf[4];
#pragma unroll
            for (int i = 0; i < 2; ++i) af[i] = frag_a(QPs, WW, wave_m * 32 + i * 16, k0, lane);
#pragma unroll
            for (int j = 0; j < 4; ++j) bf[j] = frag_b(KVs, HH, wave_n * 64 + j * 16, k0, lane);
#pragma unroll
            for (int i = 0; i < 2; ++i)
#pragma unroll
                for (int j = 0; j < 4; ++j) acc[i][j] = WMMA_BF16(af[i], bf[j], acc[i][j]);
        }
#pragma unroll
        for (int i = 0; i < 2; ++i)
#pragma unroll
            for (int j = 0; j < 4; ++j)
#pragma unroll
                for (int r = 0; r < 8; ++r) {
                    int m = wave_m * 32 + i * 16 + r + mhi;
                    int n = wave_n * 64 + j * 16 + ncol;
                    V[base + m * WW + n] = f2bf(acc[i][j][r]);
                }
    }
}

// ---------------------------------------------------------------------------
// Kernel 3a: repack 3x3 weights OIHW fp32 -> [kh*3+kw][oc][ic] bf16
// ---------------------------------------------------------------------------
__global__ void repack_w_kernel(const float* __restrict__ w, __bf16* __restrict__ wbf) {
    int i = blockIdx.x * 256 + threadIdx.x;            // over 256*256*9
    if (i >= CCH * CCH * 9) return;
    int khw = i % 9, rest = i / 9;
    int ic = rest % CCH, oc = rest / CCH;
    wbf[(size_t)khw * CCH * CCH + (size_t)oc * CCH + ic] = f2bf(w[i]);
}

__global__ void zero_stats_kernel(float* gsum, float* gsumsq) {
    int t = threadIdx.x;
    if (t < CCH) { gsum[t] = 0.f; gsumsq[t] = 0.f; }
}

// ---------------------------------------------------------------------------
// Kernel 3b: 3x3 conv (pad 1) as implicit GEMM + bias + BN partial sums.
// grid (B*H, C/128), block 256; wave = 64(M=oc) x 32(N=w).
// A tile (pre-packed bf16 weights) is fetched with async global->LDS.
// ---------------------------------------------------------------------------
__global__ __launch_bounds__(256) void conv3x3_kernel(
    const __bf16* __restrict__ att, const __bf16* __restrict__ wbf,
    const float* __restrict__ b_out, float* __restrict__ o,
    float* __restrict__ gsum, float* __restrict__ gsumsq) {
    __shared__ __align__(16) __bf16 As[128 * 32];
    __shared__ __align__(16) __bf16 Bs[128 * 32];
    __shared__ float bnsum[128];
    __shared__ float bnsq[128];

    const int tid = threadIdx.x, lane = tid & 31, wid = tid >> 5;
    const int wave_m = wid >> 2, wave_n = wid & 3;          // 2 x 4 wave grid
    const int b = blockIdx.x >> 7, h = blockIdx.x & 127;
    const int oc0 = blockIdx.y * 128;

    v8f acc[4][2];
#pragma unroll
    for (int i = 0; i < 4; ++i)
#pragma unroll
        for (int j = 0; j < 2; ++j) acc[i][j] = (v8f)V8F_ZERO;

    for (int kh = 0; kh < 3; ++kh) {
        const int ih = h + kh - 1;
        const bool hok = (ih >= 0) && (ih < HH);
        for (int kw = 0; kw < 3; ++kw) {
            const __bf16* ws = wbf + (size_t)(kh * 3 + kw) * CCH * CCH;
            for (int icb = 0; icb < CCH / 32; ++icb) {
                const int ic0 = icb * 32;
                // A: 128(oc) x 32(ic) bf16 -- async byte copy to LDS
#pragma unroll
                for (int t = tid; t < 512; t += 256) {
                    int row = t >> 2, kk8 = (t & 3) * 8;
                    async_copy_b128(&ws[(size_t)(oc0 + row) * CCH + ic0 + kk8],
                                    &As[row * 32 + kk8]);
                }
                // B: att[b][ic][ih][w+kw-1] (bf16) -> LDS [n=w][k=ic], 0-padded
#pragma unroll
                for (int t = tid; t < 4096; t += 256) {
                    int kk = t >> 7, n = t & 127;
                    int iw = n + kw - 1;
                    __bf16 v = (__bf16)0.0f;
                    if (hok && iw >= 0 && iw < WW)
                        v = att[(((size_t)b * CCH + ic0 + kk) * HH + ih) * WW + iw];
                    Bs[n * 32 + kk] = v;
                }
                wait_async();
                __syncthreads();

                v16bf af[4], bf[2];
#pragma unroll
                for (int i = 0; i < 4; ++i) af[i] = frag_a(As, 32, wave_m * 64 + i * 16, 0, lane);
#pragma unroll
                for (int j = 0; j < 2; ++j) bf[j] = frag_b(Bs, 32, wave_n * 32 + j * 16, 0, lane);
#pragma unroll
                for (int i = 0; i < 4; ++i)
#pragma unroll
                    for (int j = 0; j < 2; ++j) acc[i][j] = WMMA_BF16(af[i], bf[j], acc[i][j]);
                __syncthreads();
            }
        }
    }

    if (tid < 128) { bnsum[tid] = 0.f; bnsq[tid] = 0.f; }
    __syncthreads();

    const int mhi = (lane >> 4) * 8, ncol = lane & 15;
#pragma unroll
    for (int i = 0; i < 4; ++i)
#pragma unroll
        for (int j = 0; j < 2; ++j)
#pragma unroll
            for (int r = 0; r < 8; ++r) {
                int m = wave_m * 64 + i * 16 + r + mhi;
                int n = wave_n * 32 + j * 16 + ncol;
                int oc = oc0 + m;
                float v = acc[i][j][r] + b_out[oc];
                o[(((size_t)b * CCH + oc) * HH + h) * WW + n] = v;
                atomicAdd(&bnsum[m], v);
                atomicAdd(&bnsq[m], v * v);
            }
    __syncthreads();
    if (tid < 128) {
        atomicAdd(&gsum[oc0 + tid],   bnsum[tid]);
        atomicAdd(&gsumsq[oc0 + tid], bnsq[tid]);
    }
}

// ---------------------------------------------------------------------------
// Kernel 4: BatchNorm (batch stats) + LeakyReLU, in place on d_out, float4.
// ---------------------------------------------------------------------------
__global__ __launch_bounds__(256) void bn_leaky_kernel(
    float* __restrict__ o, const float* __restrict__ gsum,
    const float* __restrict__ gsumsq, const float* __restrict__ gamma,
    const float* __restrict__ beta) {
    size_t i4 = (size_t)blockIdx.x * 256 + threadIdx.x;
    size_t base = i4 * 4;
    if (base >= NELEM) return;
    int c = (int)((base >> 14) & (CCH - 1));   // layout [b][c][16384]
    const float inv_n = 1.f / (float)(BATCH * HWSZ);
    float mean = gsum[c] * inv_n;
    float var  = gsumsq[c] * inv_n - mean * mean;
    float rs   = rsqrtf(var + BN_EPS);
    float g    = gamma[c] * rs;
    float bt   = beta[c] - mean * g;
    float4 v = *reinterpret_cast<float4*>(&o[base]);
    float a0 = v.x * g + bt, a1 = v.y * g + bt, a2 = v.z * g + bt, a3 = v.w * g + bt;
    v.x = a0 > 0.f ? a0 : LEAKY * a0;
    v.y = a1 > 0.f ? a1 : LEAKY * a1;
    v.z = a2 > 0.f ? a2 : LEAKY * a2;
    v.w = a3 > 0.f ? a3 : LEAKY * a3;
    *reinterpret_cast<float4*>(&o[base]) = v;
}

// ---------------------------------------------------------------------------
// Launcher.  Workspace layout (requires ~203 MB):
//   [0)          q buffer   NELEM bf16
//   [+NELEM)     k buffer   NELEM bf16
//   [+2*NELEM)   v buffer   NELEM bf16  (overwritten by att)
//   [+3*NELEM)   repacked weights 9*256*256 bf16
//   [+...]       gsum[256], gsumsq[256] f32
// Conv output lives in d_out (f32); bn_leaky normalizes it in place.
// ---------------------------------------------------------------------------
extern "C" void kernel_launch(void* const* d_in, const int* in_sizes, int n_in,
                              void* d_out, int out_size, void* d_ws, size_t ws_size,
                              hipStream_t stream) {
    const float* x     = (const float*)d_in[0];
    const float* y     = (const float*)d_in[1];
    const float* z     = (const float*)d_in[2];
    const float* wq    = (const float*)d_in[3];
    const float* bq    = (const float*)d_in[4];
    const float* wk    = (const float*)d_in[5];
    const float* bk    = (const float*)d_in[6];
    const float* wv    = (const float*)d_in[7];
    const float* bv    = (const float*)d_in[8];
    const float* w_out = (const float*)d_in[9];
    const float* b_out = (const float*)d_in[10];
    const float* gamma = (const float*)d_in[11];
    const float* beta  = (const float*)d_in[12];
    float* out = (float*)d_out;

    __bf16* qbuf = (__bf16*)d_ws;
    __bf16* kbuf = qbuf + NELEM;
    __bf16* vbuf = kbuf + NELEM;
    __bf16* wbf  = vbuf + NELEM;
    float* gsum   = (float*)(wbf + (size_t)9 * CCH * CCH);
    float* gsumsq = gsum + CCH;

    dim3 blk(256);

    // 1x1 projections q, k, v (fp32 -> bf16)
    dim3 gproj(HWSZ / 128, CCH / 128, BATCH);
    proj1x1_kernel<<<gproj, blk, 0, stream>>>(x, wq, bq, qbuf);
    proj1x1_kernel<<<gproj, blk, 0, stream>>>(y, wk, bk, kbuf);
    proj1x1_kernel<<<gproj, blk, 0, stream>>>(z, wv, bv, vbuf);

    // attention per (b, c); att overwrites vbuf
    attention_kernel<<<dim3(CCH, BATCH), blk, 0, stream>>>(qbuf, kbuf, vbuf);

    // weight repack + stat zeroing
    repack_w_kernel<<<(CCH * CCH * 9 + 255) / 256, blk, 0, stream>>>(w_out, wbf);
    zero_stats_kernel<<<1, blk, 0, stream>>>(gsum, gsumsq);

    // 3x3 conv -> d_out (+ per-channel sums)
    conv3x3_kernel<<<dim3(BATCH * HH, CCH / 128), blk, 0, stream>>>(
        vbuf, wbf, b_out, out, gsum, gsumsq);

    // BN + LeakyReLU in place
    bn_leaky_kernel<<<(unsigned)((NELEM / 4 + 255) / 256), blk, 0, stream>>>(
        out, gsum, gsumsq, gamma, beta);
}
